// AttMultiHeadDot_87832081203819
// MI455X (gfx1250) — matrix operand order repaired and checked by reading, not compiled, not run on hardware
//
#include <hip/hip_runtime.h>
#include <hip/hip_bf16.h>
#include <math.h>

// Problem constants (from reference)
#define B_   32
#define T_   1500
#define D_   1024        // EPROJS == DUNITS
#define H_   8
#define DK_  128
#define DV_  128
#define SCALING 0.08838834764831845f   // 1/sqrt(128)

typedef _Float16 f16;
typedef __attribute__((ext_vector_type(16))) _Float16 v16h;
typedef __attribute__((ext_vector_type(8)))  _Float16 v8h;
typedef __attribute__((ext_vector_type(8)))  float    v8f;
typedef __attribute__((ext_vector_type(4)))  float    v4f;

// ---------------------------------------------------------------------------
// Prep: convert Wk/Wv (f32 [h][d][n]) -> f16 transposed [h][n][k] so that the
// WMMA B-fragment (lane = N column, K contiguous) is two 16B loads.
// Also zero the context accumulator c.
// ---------------------------------------------------------------------------
__global__ __launch_bounds__(256)
void prep_convert_kernel(const float* __restrict__ Wk, const float* __restrict__ Wv,
                         f16* __restrict__ wk16, f16* __restrict__ wv16,
                         float* __restrict__ c) {
  int idx = blockIdx.x * blockDim.x + threadIdx.x;
  if (idx < B_ * D_) c[idx] = 0.0f;
  const int total = H_ * DK_ * D_;           // 1,048,576 elements per weight
  for (int i = idx; i < total; i += gridDim.x * blockDim.x) {
    int h = i >> 17;                          // / (128*1024)
    int r = i & ((1 << 17) - 1);
    int n = r >> 10;                          // / 1024
    int k = r & 1023;
    // src layout [h][k][n], dst layout [h][n][k]
    wk16[i] = (f16)Wk[((size_t)h * D_ + k) * DK_ + n];
    wv16[i] = (f16)Wv[((size_t)h * D_ + k) * DK_ + n];
  }
}

// ---------------------------------------------------------------------------
// q[b, h*128+kk] = tanh(dec_z[b,:] . Wq[h,:,kk] + bq[h,kk])   (tiny: 67 MFLOP)
// ---------------------------------------------------------------------------
__global__ __launch_bounds__(256)
void prep_q_kernel(const float* __restrict__ dec_z, const float* __restrict__ Wq,
                   const float* __restrict__ bq, float* __restrict__ q) {
  int idx = blockIdx.x * 256 + threadIdx.x;   // 32768 = B*1024
  int b = idx >> 10;
  int n = idx & 1023;
  int h = n >> 7;
  int kk = n & 127;
  float s = bq[h * DK_ + kk];
  const float* z  = dec_z + (size_t)b * D_;
  const float* wq = Wq + (size_t)h * D_ * DK_ + kk;
  for (int d = 0; d < D_; ++d) s += z[d] * wq[(size_t)d * DK_];
  q[idx] = tanhf(s);
}

// ---------------------------------------------------------------------------
// score: e[b,h,t] = tanh(enc[b,t,:] @ Wk_h) . q[b,h,:]
// Workgroup = (16 t-rows, batch b); 8 waves, wave == head.
// Each wave: 16x128x1024 GEMM via V_WMMA_F32_16X16X32_F16, 8 acc tiles.
// ---------------------------------------------------------------------------
__global__ __launch_bounds__(256)
void score_kernel(const float* __restrict__ enc, const f16* __restrict__ wk16,
                  const float* __restrict__ q, float* __restrict__ e) {
  const int b    = blockIdx.y;
  const int t0   = blockIdx.x * 16;
  const int lane = threadIdx.x & 31;
  const int h    = threadIdx.x >> 5;          // wave index == head
  const bool hi  = lane >= 16;
  const int l15  = lane & 15;

  const float* encb = enc + (size_t)b * T_ * D_;
  int rowA = t0 + l15;
  if (rowA >= T_) rowA = T_ - 1;              // clamp; out-of-range rows not written
  const float* arow = encb + (size_t)rowA * D_;
  const int kA0 = hi ? 16 : 0;                // A layout: elems0-15 K=k0+kA0..k0+kA0+15
  const f16* wh = wk16 + (size_t)h * DK_ * D_; // [n][k]
  const int kB0 = hi ? 16 : 0;                // B layout: lanes16-31 hold K=16..31

  const v8f vzero = {0.f, 0.f, 0.f, 0.f, 0.f, 0.f, 0.f, 0.f};
  v8f acc[8];
#pragma unroll
  for (int j = 0; j < 8; ++j) acc[j] = vzero;

  for (int k0 = 0; k0 < D_; k0 += 32) {
    // A fragment: 16 f32 -> f16 per lane (on-the-fly convert, no LDS needed;
    // the 8 waves share these lines through the WGP cache)
    v4f f0 = *(const v4f*)(arow + k0 + kA0);
    v4f f1 = *(const v4f*)(arow + k0 + kA0 + 4);
    v4f f2 = *(const v4f*)(arow + k0 + kA0 + 8);
    v4f f3 = *(const v4f*)(arow + k0 + kA0 + 12);
    v16h a;
#pragma unroll
    for (int i = 0; i < 4; ++i) {
      a[i]      = (f16)f0[i];
      a[4 + i]  = (f16)f1[i];
      a[8 + i]  = (f16)f2[i];
      a[12 + i] = (f16)f3[i];
    }
    if (k0 + 32 < D_)
      __builtin_prefetch(wh + (size_t)l15 * D_ + k0 + 32 + kB0, 0, 1);
#pragma unroll
    for (int j = 0; j < 8; ++j) {
      const f16* bp = wh + (size_t)(j * 16 + l15) * D_ + k0 + kB0;
      v8h b0 = *(const v8h*)bp;
      v8h b1 = *(const v8h*)(bp + 8);
      v16h bb = __builtin_shufflevector(b0, b1, 0, 1, 2, 3, 4, 5, 6, 7,
                                        8, 9, 10, 11, 12, 13, 14, 15);
      acc[j] = __builtin_amdgcn_wmma_f32_16x16x32_f16(
          false, a, false, bb, (short)0, acc[j], false, false);
    }
  }

  // e[t,h] = sum_n tanh(acc[m][n]) * q[b,h,n]
  // C layout: VGPR r -> (M=r, N=lane) for lanes 0-15; (M=8+r, N=lane-16) for 16-31.
  const float* qb = q + (size_t)b * (H_ * DK_) + h * DK_;
  float partial[8];
#pragma unroll
  for (int r = 0; r < 8; ++r) partial[r] = 0.f;
#pragma unroll
  for (int j = 0; j < 8; ++j) {
    float qv = qb[j * 16 + l15];
#pragma unroll
    for (int r = 0; r < 8; ++r) partial[r] += tanhf(acc[j][r]) * qv;
  }
#pragma unroll
  for (int r = 0; r < 8; ++r) {   // reduce across the 16 lanes sharing an M row
    partial[r] += __shfl_xor(partial[r], 1);
    partial[r] += __shfl_xor(partial[r], 2);
    partial[r] += __shfl_xor(partial[r], 4);
    partial[r] += __shfl_xor(partial[r], 8);
  }
  float* eb = e + ((size_t)b * H_ + h) * T_;
  if (l15 == 0) {                  // lane 0 writes rows t0..t0+7, lane 16 rows t0+8..t0+15
    int mbase = t0 + (hi ? 8 : 0);
#pragma unroll
    for (int r = 0; r < 8; ++r)
      if (mbase + r < T_) eb[mbase + r] = partial[r];
  }
}

// ---------------------------------------------------------------------------
// Masked softmax over T per (b,h): w = softmax(SCALING * e with t>=len -> -inf)
// ---------------------------------------------------------------------------
__global__ __launch_bounds__(256)
void softmax_kernel(const float* __restrict__ e, const long long* __restrict__ lens,
                    float* __restrict__ w_ws, float* __restrict__ w_out) {
  const int bh = blockIdx.x;
  const int b  = bh / H_;
  const float* eb = e + (size_t)bh * T_;
  const int len = (int)lens[b];
  __shared__ float red[256];
  const int tid = threadIdx.x;

  float m = -INFINITY;
  for (int t = tid; t < len; t += 256) m = fmaxf(m, eb[t]);
  red[tid] = m; __syncthreads();
  for (int s = 128; s > 0; s >>= 1) {
    if (tid < s) red[tid] = fmaxf(red[tid], red[tid + s]);
    __syncthreads();
  }
  m = red[0];
  __syncthreads();

  float sum = 0.f;
  for (int t = tid; t < len; t += 256) sum += expf(SCALING * (eb[t] - m));
  red[tid] = sum; __syncthreads();
  for (int s = 128; s > 0; s >>= 1) {
    if (tid < s) red[tid] += red[tid + s];
    __syncthreads();
  }
  sum = red[0];
  const float inv = 1.0f / sum;

  for (int t = tid; t < T_; t += 256) {
    float wv = (t < len) ? expf(SCALING * (eb[t] - m)) * inv : 0.0f;
    w_ws[(size_t)bh * T_ + t]  = wv;
    w_out[(size_t)bh * T_ + t] = wv;
  }
}

// ---------------------------------------------------------------------------
// context: c[b,h,:] += sum_{t in tile} w[b,h,t] * (enc[b,t,:] @ Wv_h)
// Same fused WMMA GEMM; v is never materialized.
// ---------------------------------------------------------------------------
__global__ __launch_bounds__(256)
void context_kernel(const float* __restrict__ enc, const f16* __restrict__ wv16,
                    const float* __restrict__ w, float* __restrict__ c) {
  const int b    = blockIdx.y;
  const int t0   = blockIdx.x * 16;
  const int lane = threadIdx.x & 31;
  const int h    = threadIdx.x >> 5;
  const bool hi  = lane >= 16;
  const int l15  = lane & 15;

  const float* encb = enc + (size_t)b * T_ * D_;
  int rowA = t0 + l15;
  if (rowA >= T_) rowA = T_ - 1;
  const float* arow = encb + (size_t)rowA * D_;
  const int kA0 = hi ? 16 : 0;
  const f16* wh = wv16 + (size_t)h * DK_ * D_;
  const int kB0 = hi ? 16 : 0;

  // attention weights for this lane's 8 M rows (0 beyond T -> masked rows contribute 0)
  const float* wb = w + ((size_t)b * H_ + h) * T_;
  const int mb = t0 + (hi ? 8 : 0);
  float wr[8];
#pragma unroll
  for (int r = 0; r < 8; ++r) wr[r] = (mb + r < T_) ? wb[mb + r] : 0.0f;

  const v8f vzero = {0.f, 0.f, 0.f, 0.f, 0.f, 0.f, 0.f, 0.f};
  v8f acc[8];
#pragma unroll
  for (int j = 0; j < 8; ++j) acc[j] = vzero;

  for (int k0 = 0; k0 < D_; k0 += 32) {
    v4f f0 = *(const v4f*)(arow + k0 + kA0);
    v4f f1 = *(const v4f*)(arow + k0 + kA0 + 4);
    v4f f2 = *(const v4f*)(arow + k0 + kA0 + 8);
    v4f f3 = *(const v4f*)(arow + k0 + kA0 + 12);
    v16h a;
#pragma unroll
    for (int i = 0; i < 4; ++i) {
      a[i]      = (f16)f0[i];
      a[4 + i]  = (f16)f1[i];
      a[8 + i]  = (f16)f2[i];
      a[12 + i] = (f16)f3[i];
    }
    if (k0 + 32 < D_)
      __builtin_prefetch(wh + (size_t)l15 * D_ + k0 + 32 + kB0, 0, 1);
#pragma unroll
    for (int j = 0; j < 8; ++j) {
      const f16* bp = wh + (size_t)(j * 16 + l15) * D_ + k0 + kB0;
      v8h b0 = *(const v8h*)bp;
      v8h b1 = *(const v8h*)(bp + 8);
      v16h bb = __builtin_shufflevector(b0, b1, 0, 1, 2, 3, 4, 5, 6, 7,
                                        8, 9, 10, 11, 12, 13, 14, 15);
      acc[j] = __builtin_amdgcn_wmma_f32_16x16x32_f16(
          false, a, false, bb, (short)0, acc[j], false, false);
    }
  }

  // c[n] partial = sum_m w[m] * v[m,n] within this 16-row tile; one atomic per column.
#pragma unroll
  for (int j = 0; j < 8; ++j) {
    float s = 0.f;
#pragma unroll
    for (int r = 0; r < 8; ++r) s += acc[j][r] * wr[r];
    s += __shfl_xor(s, 16);          // add the other half's 8 rows (same N)
    if (!hi) atomicAdd(&c[(size_t)b * D_ + h * DK_ + j * 16 + l15], s);
  }
}

// ---------------------------------------------------------------------------
// out[b,:] = c[b,:] @ Wo   (32x1024x1024, 67 MFLOP — scalar is fine)
// ---------------------------------------------------------------------------
__global__ __launch_bounds__(256)
void out_kernel(const float* __restrict__ c, const float* __restrict__ Wo,
                float* __restrict__ out) {
  int idx = blockIdx.x * 256 + threadIdx.x;   // 32768 = B*1024
  int b = idx >> 10;
  int o = idx & 1023;
  const float* cb = c + (size_t)b * D_;
  float s = 0.f;
  for (int n = 0; n < D_; ++n) s += cb[n] * Wo[(size_t)n * D_ + o];
  out[idx] = s;
}

// ---------------------------------------------------------------------------
extern "C" void kernel_launch(void* const* d_in, const int* in_sizes, int n_in,
                              void* d_out, int out_size, void* d_ws, size_t ws_size,
                              hipStream_t stream) {
  const float*     enc   = (const float*)d_in[0];      // (B,T,1024) f32
  const long long* lens  = (const long long*)d_in[1];  // (B,) int64
  const float*     dec_z = (const float*)d_in[2];      // (B,1024)
  const float*     Wq    = (const float*)d_in[3];      // (H,1024,128)
  const float*     bq    = (const float*)d_in[4];      // (H,128)
  const float*     Wk    = (const float*)d_in[5];      // (H,1024,128)
  const float*     Wv    = (const float*)d_in[6];      // (H,1024,128)
  const float*     Wo    = (const float*)d_in[7];      // (1024,1024)

  float* out   = (float*)d_out;            // first output: (B,1024)
  float* w_out = out + B_ * D_;            // second output: (B,H,T)

  // workspace carve-out (~7.5 MB)
  char* p = (char*)d_ws;
  float* q   = (float*)p;  p += (size_t)B_ * D_ * sizeof(float);
  float* c   = (float*)p;  p += (size_t)B_ * D_ * sizeof(float);
  float* e   = (float*)p;  p += (size_t)B_ * H_ * T_ * sizeof(float);
  float* wts = (float*)p;  p += (size_t)B_ * H_ * T_ * sizeof(float);
  f16*  wk16 = (f16*)p;    p += (size_t)H_ * DK_ * D_ * sizeof(f16);
  f16*  wv16 = (f16*)p;

  prep_convert_kernel<<<1024, 256, 0, stream>>>(Wk, Wv, wk16, wv16, c);
  prep_q_kernel<<<(B_ * D_) / 256, 256, 0, stream>>>(dec_z, Wq, bq, q);

  dim3 gtile((T_ + 15) / 16, B_);
  score_kernel<<<gtile, 256, 0, stream>>>(enc, wk16, q, e);
  softmax_kernel<<<B_ * H_, 256, 0, stream>>>(e, lens, wts, w_out);
  context_kernel<<<gtile, 256, 0, stream>>>(enc, wv16, wts, c);
  out_kernel<<<(B_ * D_) / 256, 256, 0, stream>>>(c, Wo, out);
}
